// MTIF_26637387170393
// MI455X (gfx1250) — matrix operand order repaired and checked by reading, not compiled, not run on hardware
//
#include <hip/hip_runtime.h>
#include <hip/hip_bf16.h>

// ---------------- problem constants ----------------
#define N_NODES 20000
#define N_EDGES 160000
#define IN_DIM  2000
#define ED      256      // embedding dim (output of fused EI gemm)
#define ED2     512      // inner dim of fused weight product
#define HID     128
#define GH      16
#define LAYERS  4
#define NC      2
#define KBLK    63       // ceil(2000/32) K-blocks of 32
#define MT      1250     // 20000/16 M-tiles
#define NTILES  16       // 256/16 N-tiles
#define BN_EPS  1e-5f

typedef __attribute__((ext_vector_type(16))) _Float16 v16h;
typedef __attribute__((ext_vector_type(8)))  float    v8f;
typedef __attribute__((ext_vector_type(4)))  unsigned v4u;

// ---------------- CDNA5 async global->LDS helpers (ASYNCcnt path) ----------
__device__ __forceinline__ void async_b128(unsigned lds_off, const void* gptr)
{
    asm volatile("global_load_async_to_lds_b128 %0, %1, off"
                 :: "v"(lds_off), "v"((unsigned long long)(size_t)gptr)
                 : "memory");
}
__device__ __forceinline__ void wait_async0()
{
    asm volatile("s_wait_asynccnt 0x0" ::: "memory");
}

__device__ __forceinline__ v16h make_afrag(const float* a, const float* b)
{
    v16h f;
#pragma unroll
    for (int i = 0; i < 8; ++i) { f[i] = (_Float16)a[i]; f[8 + i] = (_Float16)b[i]; }
    return f;
}

// ---------------------------------------------------------------------------
// K0: fuse W_ei1 @ W_ei2 -> f16, pre-swizzled into WMMA B-fragment layout.
//   lane<16 : dword v holds K = 2v, 2v+1 (v<4) / 8+2v, 9+2v (v>=4)   (K 0..7, 16..23)
//   lane>=16: dword v holds K = 8+2v, 9+2v (v<4) / 16+2v, 17+2v (v>=4) (K 8..15, 24..31)
//   N = nt*16 + (lane & 15)
// ---------------------------------------------------------------------------
__global__ void pack_b_kernel(const float* __restrict__ W1,
                              const float* __restrict__ W2,
                              unsigned* __restrict__ Bp)
{
    int idx = blockIdx.x * blockDim.x + threadIdx.x;
    if (idx >= KBLK * NTILES * 32 * 8) return;
    int v    = idx & 7;
    int lane = (idx >> 3) & 31;
    int nt   = (idx >> 8) & 15;
    int kb   = idx >> 12;
    int n    = nt * 16 + (lane & 15);
    int kk   = 2 * v + ((lane < 16) ? 0 : 8) + ((v < 4) ? 0 : 8);
    int k0   = kb * 32 + kk;
    float a0 = 0.f, a1 = 0.f;
    if (k0 < IN_DIM) {
        const float* r = W1 + k0 * ED2;
        for (int m = 0; m < ED2; ++m) a0 = fmaf(r[m], W2[m * ED + n], a0);
    }
    if (k0 + 1 < IN_DIM) {
        const float* r = W1 + (k0 + 1) * ED2;
        for (int m = 0; m < ED2; ++m) a1 = fmaf(r[m], W2[m * ED + n], a1);
    }
    union { _Float16 h[2]; unsigned u; } p;
    p.h[0] = (_Float16)a0;
    p.h[1] = (_Float16)a1;
    Bp[idx] = p.u;
}

// fused bias: b = b_ei1 @ W_ei2 + b_ei2
__global__ void bfuse_kernel(const float* __restrict__ b1,
                             const float* __restrict__ W2,
                             const float* __restrict__ b2,
                             float* __restrict__ bf)
{
    int n = threadIdx.x;
    if (n >= ED) return;
    float a = b2[n];
    for (int m = 0; m < ED2; ++m) a = fmaf(b1[m], W2[m * ED + n], a);
    bf[n] = a;
}

// pack Wp2 [128x128] into B-fragment layout: [kb(4)][nt(8)][lane(32)][v(8)]
__global__ void pack_wp2_kernel(const float* __restrict__ Wp2,
                                unsigned* __restrict__ out)
{
    int idx = blockIdx.x * blockDim.x + threadIdx.x;
    if (idx >= 4 * 8 * 32 * 8) return;
    int v    = idx & 7;
    int lane = (idx >> 3) & 31;
    int nt   = (idx >> 8) & 7;
    int kb   = idx >> 11;
    int n    = nt * 16 + (lane & 15);
    int kk   = 2 * v + ((lane < 16) ? 0 : 8) + ((v < 4) ? 0 : 8);
    int k    = kb * 32 + kk;
    union { _Float16 h[2]; unsigned u; } p;
    p.h[0] = (_Float16)Wp2[k * HID + n];
    p.h[1] = (_Float16)Wp2[(k + 1) * HID + n];
    out[idx] = p.u;
}

// ---------------------------------------------------------------------------
// K1: ei = X @ Wfused + bfused via V_WMMA_F32_16X16X32_F16.
// One wave per 16-row M-tile, owning all 16 N-tiles. B is staged through LDS
// with double-buffered global_load_async_to_lds_b128 (shared by all 5 waves).
// ---------------------------------------------------------------------------
__global__ void __launch_bounds__(160)
gemm_ei_kernel(const float* __restrict__ X,
               const unsigned* __restrict__ Bp,
               const float* __restrict__ bfuse,
               float* __restrict__ ei)
{
    __shared__ v4u sB[2][1024];                  // 2 x 16KB B-block buffers
    const int tid  = threadIdx.x;
    const int lane = tid & 31;
    const int wv   = tid >> 5;                   // 0..4
    const int mtile = blockIdx.x * 5 + wv;       // grid=250 -> always < 1250
    const int sel  = lane >> 4;
    const int rloc = lane & 15;
    const float* xrow = X + (mtile * 16 + rloc) * IN_DIM;

    // prologue: async-fill buffer 0 with kb=0 fragments
    for (int c = tid; c < 1024; c += 160)
        async_b128((unsigned)(size_t)&sB[0][c], Bp + c * 4);

    float xa[8], xb[8];
    {
        int baseA = sel * 8, baseB = baseA + 16; // kb=0 always in range
#pragma unroll
        for (int i = 0; i < 8; ++i) { xa[i] = xrow[baseA + i]; xb[i] = xrow[baseB + i]; }
    }
    v16h afrag = make_afrag(xa, xb);

    v8f acc[NTILES];
#pragma unroll
    for (int nt = 0; nt < NTILES; ++nt) acc[nt] = (v8f){};

    wait_async0();
    __syncthreads();

    for (int kb = 0; kb < KBLK; ++kb) {
        const int cur = kb & 1;
        const int nb  = kb + 1;
        // kick off async fill of the other buffer with the next K-block
        if (nb < KBLK) {
            const unsigned* gb = Bp + (size_t)nb * 4096;
            for (int c = tid; c < 1024; c += 160)
                async_b128((unsigned)(size_t)&sB[cur ^ 1][c], gb + c * 4);
        }
        // prefetch next A chunk (global, zero-padded K tail)
        float nxa[8], nxb[8];
        if (nb < KBLK) {
            int baseA = nb * 32 + sel * 8, baseB = baseA + 16;
            if (baseA < IN_DIM) {
#pragma unroll
                for (int i = 0; i < 8; ++i) nxa[i] = xrow[baseA + i];
            } else {
#pragma unroll
                for (int i = 0; i < 8; ++i) nxa[i] = 0.f;
            }
            if (baseB < IN_DIM) {
#pragma unroll
                for (int i = 0; i < 8; ++i) nxb[i] = xrow[baseB + i];
            } else {
#pragma unroll
                for (int i = 0; i < 8; ++i) nxb[i] = 0.f;
            }
        }
        // compute on current buffer; depth-1 LDS read pipeline
        union { v4u u[2]; v16h h; } bcur, bnxt;
        {
            const v4u* lb = &sB[cur][(size_t)lane * 2];
            bcur.u[0] = lb[0]; bcur.u[1] = lb[1];
        }
#pragma unroll
        for (int nt = 0; nt < NTILES; ++nt) {
            if (nt + 1 < NTILES) {
                const v4u* ln = &sB[cur][(size_t)((nt + 1) * 32 + lane) * 2];
                bnxt.u[0] = ln[0]; bnxt.u[1] = ln[1];
            }
            acc[nt] = __builtin_amdgcn_wmma_f32_16x16x32_f16(
                false, afrag, false, bcur.h, (short)0, acc[nt], false, false);
            bcur = bnxt;
        }
        wait_async0();       // next buffer landed in LDS
        __syncthreads();     // everyone done reading current buffer
        if (nb < KBLK) afrag = make_afrag(nxa, nxb);
    }

    int col = lane & 15;
#pragma unroll
    for (int nt = 0; nt < NTILES; ++nt) {
        float bf = bfuse[nt * 16 + col];
#pragma unroll
        for (int r = 0; r < 8; ++r) {
            int m = mtile * 16 + r + 8 * sel;    // C/D: lanes16-31 -> M = r+8
            ei[m * ED + nt * 16 + col] = acc[nt][r] + bf;
        }
    }
}

// ---------------------------------------------------------------------------
// K2: PEWE edge weights with WMMA. One wave owns 16 edges.
// Hidden vectors are computed directly in A-fragment layout; hh = H @ Wp2
// via 4xK WMMA per 16-col tile; cosine terms reduced per 16-lane half.
// ---------------------------------------------------------------------------
__global__ void __launch_bounds__(256)
edgew_wmma_kernel(const float* __restrict__ ni, const int* __restrict__ eidx,
                  const float* __restrict__ ei, const unsigned* __restrict__ Bwp,
                  const float* __restrict__ Wp1, const float* __restrict__ bp1,
                  const float* __restrict__ bn1g, const float* __restrict__ bn1b,
                  const float* __restrict__ bp2, float* __restrict__ ew)
{
    __shared__ v4u  sBw[2048];                   // packed Wp2 fragments, 32KB
    __shared__ float sW1a[HID], sW1b[HID], sBp1[HID], sS1[HID], sBb[HID], sBp2[HID];
    const float inv = rsqrtf(1.f + BN_EPS);
    {
        const v4u* g = (const v4u*)Bwp;
        for (int i = threadIdx.x; i < 2048; i += 256) sBw[i] = g[i];
        for (int i = threadIdx.x; i < HID; i += 256) {
            sW1a[i] = Wp1[i];
            sW1b[i] = Wp1[HID + i];
            sBp1[i] = bp1[i];
            sS1[i]  = bn1g[i] * inv;
            sBb[i]  = bn1b[i];
            sBp2[i] = bp2[i];
        }
    }
    __syncthreads();

    const int lane = threadIdx.x & 31;
    const int job  = (int)((blockIdx.x * blockDim.x + threadIdx.x) >> 5); // < 10000
    const int sel  = lane >> 4;
    const int erow = lane & 15;
    const int e0   = job * 16;
    const int eA   = e0 + erow;                  // A-matrix row -> edge
    const float x0 = ni[4 * eA + 0], x1 = ni[4 * eA + 1];
    const float y0 = ni[4 * eA + 2], y1 = ni[4 * eA + 3];

    // hidden vectors in A-fragment layout (K=128 -> 4 blocks of 32)
    v16h a1[4], a2[4];
#pragma unroll
    for (int kb = 0; kb < 4; ++kb) {
#pragma unroll
        for (int i = 0; i < 16; ++i) {
            int k = kb * 32 + ((i < 8) ? i : i + 8) + 8 * sel;
            float ha = fmaf(fmaxf(fmaf(x1, sW1b[k], fmaf(x0, sW1a[k], sBp1[k])), 0.f),
                            sS1[k], sBb[k]);
            float hb = fmaf(fmaxf(fmaf(y1, sW1b[k], fmaf(y0, sW1a[k], sBp1[k])), 0.f),
                            sS1[k], sBb[k]);
            a1[kb][i] = (_Float16)ha;
            a2[kb][i] = (_Float16)hb;
        }
    }

    float pn[8], p1[8], p2[8];
#pragma unroll
    for (int r = 0; r < 8; ++r) { pn[r] = 0.f; p1[r] = 0.f; p2[r] = 0.f; }

#pragma unroll
    for (int nt = 0; nt < 8; ++nt) {
        v8f t1 = (v8f){}, t2 = (v8f){};
#pragma unroll
        for (int kb = 0; kb < 4; ++kb) {
            union { v4u u[2]; v16h h; } bb;
            const v4u* bp = &sBw[(size_t)((kb * 8 + nt) * 32 + lane) * 2];
            bb.u[0] = bp[0]; bb.u[1] = bp[1];
            t1 = __builtin_amdgcn_wmma_f32_16x16x32_f16(
                false, a1[kb], false, bb.h, (short)0, t1, false, false);
            t2 = __builtin_amdgcn_wmma_f32_16x16x32_f16(
                false, a2[kb], false, bb.h, (short)0, t2, false, false);
        }
        float bq = sBp2[nt * 16 + erow];
#pragma unroll
        for (int r = 0; r < 8; ++r) {
            float u = t1[r] + bq, v = t2[r] + bq;
            pn[r] = fmaf(u, v, pn[r]);
            p1[r] = fmaf(u, u, p1[r]);
            p2[r] = fmaf(v, v, p2[r]);
        }
    }
    // reduce over the 16 lanes of each half (masks < 16 stay in-half)
#pragma unroll
    for (int m = 1; m < 16; m <<= 1) {
#pragma unroll
        for (int r = 0; r < 8; ++r) {
            pn[r] += __shfl_xor(pn[r], m, 32);
            p1[r] += __shfl_xor(p1[r], m, 32);
            p2[r] += __shfl_xor(p2[r], m, 32);
        }
    }

    // ei-gather cosine terms: each 16-lane half handles its own edge per r
    const int* srcA = eidx;
    const int* dstA = eidx + N_EDGES;
#pragma unroll
    for (int r = 0; r < 8; ++r) {
        int e = e0 + 8 * sel + r;                // C/D: lanes16-31 hold M=r+8
        int s = srcA[e], d = dstA[e];
        const float* es = ei + s * ED;
        const float* ed = ei + d * ED;
        float xn = 0.f, xs1 = 0.f, xs2 = 0.f;
#pragma unroll
        for (int t = 0; t < 16; ++t) {
            float u = es[erow + 16 * t], v = ed[erow + 16 * t];
            xn  = fmaf(u, v, xn);
            xs1 = fmaf(u, u, xs1);
            xs2 = fmaf(v, v, xs2);
        }
#pragma unroll
        for (int m = 1; m < 16; m <<= 1) {
            xn  += __shfl_xor(xn,  m, 32);
            xs1 += __shfl_xor(xs1, m, 32);
            xs2 += __shfl_xor(xs2, m, 32);
        }
        if (erow == 0) {
            float num = pn[r] + xn;
            float n1  = p1[r] + xs1;
            float n2  = p2[r] + xs2;
            float den = fmaxf(sqrtf(n1), 1e-8f) * fmaxf(sqrtf(n2), 1e-8f);
            ew[e] = (num / den + 1.f) * 0.5f;
        }
    }
}

// ---------------------------------------------------------------------------
// Laplacian weights: deg -> dis -> lw  (lw in place of ew)
// ---------------------------------------------------------------------------
__global__ void zero_kernel(float* __restrict__ p, int n)
{
    int i = blockIdx.x * blockDim.x + threadIdx.x;
    if (i < n) p[i] = 0.f;
}

__global__ void deg_kernel(const int* __restrict__ eidx,
                           const float* __restrict__ ew, float* __restrict__ deg)
{
    int e = blockIdx.x * blockDim.x + threadIdx.x;
    if (e < N_EDGES) atomicAdd(&deg[eidx[e]], ew[e]);
}

__global__ void dis_kernel(float* __restrict__ deg, int n)
{
    int i = blockIdx.x * blockDim.x + threadIdx.x;
    if (i < n) { float v = deg[i]; deg[i] = (v > 0.f) ? rsqrtf(v) : 0.f; }
}

__global__ void lw_kernel(const int* __restrict__ eidx,
                          const float* __restrict__ dis, float* __restrict__ ew)
{
    int e = blockIdx.x * blockDim.x + threadIdx.x;
    if (e >= N_EDGES) return;
    int s = eidx[e], d = eidx[N_EDGES + e];
    ew[e] = -(dis[s] * ew[e] * dis[d]);
}

// ---------------------------------------------------------------------------
// Cheb projection: B[N,48] = x @ [W0|W1|W2]  (256-dim and 16-dim variants)
// ---------------------------------------------------------------------------
__global__ void proj256_kernel(const float* __restrict__ x,
                               const float* __restrict__ W, float* __restrict__ B)
{
    __shared__ float sW[3 * ED * GH];               // 48 KB
    for (int i = threadIdx.x; i < 3 * ED * GH; i += blockDim.x) sW[i] = W[i];
    __syncthreads();
    int idx = blockIdx.x * blockDim.x + threadIdx.x;
    if (idx >= N_NODES * 48) return;
    int c = idx % 48, n = idx / 48;
    int s = c >> 4, j = c & 15;
    const float* xr = x + n * ED;
    const float* wr = sW + s * ED * GH + j;
    float acc = 0.f;
    for (int k = 0; k < ED; ++k) acc = fmaf(xr[k], wr[k * GH], acc);
    B[n * 48 + c] = acc;
}

__global__ void proj16_kernel(const float* __restrict__ x,
                              const float* __restrict__ W, float* __restrict__ B)
{
    __shared__ float sW[3 * GH * GH];
    for (int i = threadIdx.x; i < 3 * GH * GH; i += blockDim.x) sW[i] = W[i];
    __syncthreads();
    int idx = blockIdx.x * blockDim.x + threadIdx.x;
    if (idx >= N_NODES * 48) return;
    int c = idx % 48, n = idx / 48;
    int s = c >> 4, j = c & 15;
    const float* xr = x + n * GH;
    float acc = 0.f;
#pragma unroll
    for (int k = 0; k < GH; ++k) acc = fmaf(xr[k], sW[s * 256 + k * 16 + j], acc);
    B[n * 48 + c] = acc;
}

// out[dst] += lw[e] * in[src]  (16-wide, strided input view)
__global__ void prop_kernel(const int* __restrict__ eidx, const float* __restrict__ lw,
                            const float* __restrict__ in, int instride, int inoff,
                            float* __restrict__ out)
{
    int idx = blockIdx.x * blockDim.x + threadIdx.x;
    if (idx >= N_EDGES * GH) return;
    int e = idx >> 4, dc = idx & 15;
    int s = eidx[e], d = eidx[N_EDGES + e];
    atomicAdd(&out[d * GH + dc], lw[e] * in[s * instride + inoff + dc]);
}

// R = relu(b0 + t1 + 2*v - b2)
__global__ void combine_kernel(const float* __restrict__ B48, const float* __restrict__ t1,
                               const float* __restrict__ v, float* __restrict__ R)
{
    int idx = blockIdx.x * blockDim.x + threadIdx.x;
    if (idx >= N_NODES * GH) return;
    int n = idx >> 4, dc = idx & 15;
    float r = B48[n * 48 + dc] + t1[idx] + 2.f * v[idx] - B48[n * 48 + 32 + dc];
    R[idx] = fmaxf(r, 0.f);
}

// Y = 2*(conv1d_k3_pad1(R) + b); h1 = Y (first) or h1 += Y
__global__ void conv_kernel(const float* __restrict__ R, const float* __restrict__ cw,
                            const float* __restrict__ cb, float* __restrict__ Y,
                            float* __restrict__ h1, int first)
{
    __shared__ float scw[GH * GH * 3];
    __shared__ float scb[GH];
    for (int i = threadIdx.x; i < GH * GH * 3; i += blockDim.x) scw[i] = cw[i];
    if (threadIdx.x < GH) scb[threadIdx.x] = cb[threadIdx.x];
    __syncthreads();
    int idx = blockIdx.x * blockDim.x + threadIdx.x;
    if (idx >= N_NODES * GH) return;
    int n = idx >> 4, co = idx & 15;
    float acc = scb[co];
#pragma unroll
    for (int kk = 0; kk < 3; ++kk) {
        int m = n + kk - 1;
        if ((unsigned)m < (unsigned)N_NODES) {
            const float* rr = R + m * GH;
            const float* ww = scw + co * 48 + kk;
#pragma unroll
            for (int ci = 0; ci < GH; ++ci) acc = fmaf(rr[ci], ww[ci * 3], acc);
        }
    }
    float o = 2.f * acc;
    Y[idx] = o;
    if (first) h1[idx] = o; else h1[idx] += o;
}

// ---------------------------------------------------------------------------
// Final classifier: out = (relu(jk@Wc1+bc1)*s2 + bn2b) @ Wc2 + bc2
// ---------------------------------------------------------------------------
__global__ void cls_kernel(const float* __restrict__ jk, const float* __restrict__ Wc1,
                           const float* __restrict__ bc1, const float* __restrict__ g2,
                           const float* __restrict__ b2, const float* __restrict__ Wc2,
                           const float* __restrict__ bc2, float* __restrict__ out)
{
    __shared__ float sW1[GH * 256];
    __shared__ float sB1[256], sS2[256], sBB[256], sW2a[256], sW2b[256];
    const float inv = rsqrtf(1.f + BN_EPS);
    for (int i = threadIdx.x; i < GH * 256; i += blockDim.x) sW1[i] = Wc1[i];
    for (int i = threadIdx.x; i < 256; i += blockDim.x) {
        sB1[i]  = bc1[i];
        sS2[i]  = g2[i] * inv;
        sBB[i]  = b2[i];
        sW2a[i] = Wc2[2 * i];
        sW2b[i] = Wc2[2 * i + 1];
    }
    __syncthreads();
    int n = blockIdx.x * blockDim.x + threadIdx.x;
    if (n >= N_NODES) return;
    float xr[GH];
#pragma unroll
    for (int k = 0; k < GH; ++k) xr[k] = jk[n * GH + k];
    float o0 = bc2[0], o1 = bc2[1];
    for (int j = 0; j < 256; ++j) {
        float z = sB1[j];
#pragma unroll
        for (int k = 0; k < GH; ++k) z = fmaf(xr[k], sW1[k * 256 + j], z);
        z = fmaf(fmaxf(z, 0.f), sS2[j], sBB[j]);
        o0 = fmaf(z, sW2a[j], o0);
        o1 = fmaf(z, sW2b[j], o1);
    }
    out[2 * n]     = o0;
    out[2 * n + 1] = o1;
}

// ---------------------------------------------------------------------------
extern "C" void kernel_launch(void* const* d_in, const int* in_sizes, int n_in,
                              void* d_out, int out_size, void* d_ws, size_t ws_size,
                              hipStream_t stream)
{
    const float* X     = (const float*)d_in[0];
    const int*   eidx  = (const int*)d_in[1];
    const float* ni    = (const float*)d_in[2];
    // d_in[3] edge_index1 / d_in[4] edge_weight1: dead for the forward output
    const float* W1    = (const float*)d_in[5];
    const float* b1    = (const float*)d_in[6];
    const float* W2    = (const float*)d_in[7];
    const float* b2    = (const float*)d_in[8];
    const float* Wp1   = (const float*)d_in[9];
    const float* bp1   = (const float*)d_in[10];
    const float* bn1g  = (const float*)d_in[11];
    const float* bn1b  = (const float*)d_in[12];
    const float* Wp2   = (const float*)d_in[13];
    const float* bp2   = (const float*)d_in[14];
    const float* cw0   = (const float*)d_in[15];
    const float* cwl   = (const float*)d_in[16];
    const float* convw = (const float*)d_in[17];
    const float* convb = (const float*)d_in[18];
    const float* Wc1   = (const float*)d_in[19];
    const float* bc1   = (const float*)d_in[20];
    const float* bn2g  = (const float*)d_in[21];
    const float* bn2b  = (const float*)d_in[22];
    const float* Wc2   = (const float*)d_in[23];
    const float* bc2   = (const float*)d_in[24];

    char* ws = (char*)d_ws;
    size_t off = 0;
    auto alloc = [&](size_t bytes) -> void* {
        void* p = ws + off;
        off += (bytes + 255) & ~(size_t)255;
        return p;
    };
    float*    ei    = (float*)alloc((size_t)N_NODES * ED * 4);
    unsigned* Bp    = (unsigned*)alloc((size_t)KBLK * NTILES * 32 * 8 * 4);
    unsigned* Bwp   = (unsigned*)alloc((size_t)4 * 8 * 32 * 8 * 4);
    float*    bfuse = (float*)alloc(ED * 4);
    float*    ew    = (float*)alloc((size_t)N_EDGES * 4);
    float*    deg   = (float*)alloc((size_t)N_NODES * 4);
    float*    B48   = (float*)alloc((size_t)N_NODES * 48 * 4);
    float*    t1    = (float*)alloc((size_t)N_NODES * GH * 4);
    float*    uu    = (float*)alloc((size_t)N_NODES * GH * 4);
    float*    vv    = (float*)alloc((size_t)N_NODES * GH * 4);
    float*    R     = (float*)alloc((size_t)N_NODES * GH * 4);
    float*    Y     = (float*)alloc((size_t)N_NODES * GH * 4);
    float*    h1    = (float*)alloc((size_t)N_NODES * GH * 4);

    // 1) fuse EI weights + pack Wp2 into WMMA fragment layouts
    pack_b_kernel<<<(KBLK * NTILES * 32 * 8 + 255) / 256, 256, 0, stream>>>(W1, W2, Bp);
    bfuse_kernel<<<1, 256, 0, stream>>>(b1, W2, b2, bfuse);
    pack_wp2_kernel<<<(4 * 8 * 32 * 8 + 255) / 256, 256, 0, stream>>>(Wp2, Bwp);

    // 2) ei = X @ Wf + bf   (WMMA, async double-buffered B in LDS)
    gemm_ei_kernel<<<MT / 5, 160, 0, stream>>>(X, Bp, bfuse, ei);

    // 3) PEWE edge weights (WMMA, 16 edges per wave)
    edgew_wmma_kernel<<<N_EDGES / 16 / 8, 256, 0, stream>>>(
        ni, eidx, ei, Bwp, Wp1, bp1, bn1g, bn1b, bp2, ew);

    // 4) scaled-Laplacian edge weights (in place in ew)
    zero_kernel<<<(N_NODES + 255) / 256, 256, 0, stream>>>(deg, N_NODES);
    deg_kernel<<<(N_EDGES + 255) / 256, 256, 0, stream>>>(eidx, ew, deg);
    dis_kernel<<<(N_NODES + 255) / 256, 256, 0, stream>>>(deg, N_NODES);
    lw_kernel<<<(N_EDGES + 255) / 256, 256, 0, stream>>>(eidx, deg, ew);

    // 5) Cheb + conv blocks (h-stream only; g-stream is dead code)
    const int NG = N_NODES * GH;
    const int EG = N_EDGES * GH;
    for (int l = 0; l < LAYERS; ++l) {
        if (l == 0)
            proj256_kernel<<<(N_NODES * 48 + 255) / 256, 256, 0, stream>>>(ei, cw0, B48);
        else
            proj16_kernel<<<(N_NODES * 48 + 255) / 256, 256, 0, stream>>>(
                h1, cwl + (size_t)(l - 1) * 3 * GH * GH, B48);
        zero_kernel<<<(NG + 255) / 256, 256, 0, stream>>>(t1, NG);
        zero_kernel<<<(NG + 255) / 256, 256, 0, stream>>>(uu, NG);
        zero_kernel<<<(NG + 255) / 256, 256, 0, stream>>>(vv, NG);
        prop_kernel<<<(EG + 255) / 256, 256, 0, stream>>>(eidx, ew, B48, 48, 16, t1);
        prop_kernel<<<(EG + 255) / 256, 256, 0, stream>>>(eidx, ew, B48, 48, 32, uu);
        prop_kernel<<<(EG + 255) / 256, 256, 0, stream>>>(eidx, ew, uu, 16, 0, vv);
        combine_kernel<<<(NG + 255) / 256, 256, 0, stream>>>(B48, t1, vv, R);
        conv_kernel<<<(NG + 255) / 256, 256, 0, stream>>>(
            R, convw + (size_t)l * GH * GH * 3, convb + (size_t)l * GH, Y, h1, l == 0);
    }

    // 6) classifier on jk = Y (last block output)
    cls_kernel<<<(N_NODES + 255) / 256, 256, 0, stream>>>(
        Y, Wc1, bc1, bn2g, bn2b, Wc2, bc2, (float*)d_out);
}
